// TransformerEncoderWithAttn_67697274519759
// MI455X (gfx1250) — compile-verified, matrix operand order checked by reading, not run on hardware
//
#include <hip/hip_runtime.h>
#include <hip/hip_bf16.h>
#include <math.h>

// ---------------------------------------------------------------------------
// MI455X (gfx1250) transformer encoder forward.
// Compute-bound (426 GFLOP vs ~12us of HBM traffic at 23.3 TB/s), so all
// matmuls run through v_wmma_f32_16x16x32_bf16 (bf16 in, fp32 accumulate).
// GEMM A-tiles are staged by the Tensor Data Mover with double-buffered LDS
// and two loads in flight (s_wait_tensorcnt 1); weight tiles are pipelined
// through VGPRs with native bf16 converts. LN / softmax / residuals stay fp32.
// ---------------------------------------------------------------------------

typedef __attribute__((ext_vector_type(16))) __bf16 v16bf;
typedef __attribute__((ext_vector_type(8)))  __bf16 bfx8;
typedef __attribute__((ext_vector_type(8)))  float  v8f;
typedef __attribute__((ext_vector_type(4)))  unsigned int u32x4;
typedef __attribute__((ext_vector_type(8)))  int i32x8;
typedef __attribute__((ext_vector_type(4)))  int i32x4;

#define LAYERS 6
#define NHEAD  12
#define DMODEL 768
#define DFF    3072
#define BATCH  4
#define SEQ    1024
#define RTOT   (BATCH * SEQ)   // 4096 rows
#define DHEAD  64
#define QKVW   (3 * DMODEL)    // 2304

// Native RNE conversion (lowers to gfx1250 v_cvt bf16 instructions instead of
// a bfe/add3/mov_b16 emulation chain).
static __device__ __forceinline__ __bf16 f2bf(float f) { return (__bf16)f; }

// Low 32 bits of a flat shared-memory address == LDS byte offset
// (ISA 10.2: LDS aperture lives in the high dword).
static __device__ __forceinline__ unsigned lds_off(const void* p) {
  return (unsigned)(unsigned long long)p;
}

// Tensor Data Mover: DMA a 2D bf16 tile [rows x cols] (row stride
// strideElems) from global memory into LDS, inserting 16B of padding after
// each row (padCode: 3 -> pad every 64B row, 4 -> every 128B row), i.e.
// LDS pitch = cols + 8 bf16. Issued by one wave; tracked with TENSORcnt.
static __device__ __forceinline__ void tdm_load_2d(const __bf16* gsrc,
                                                   unsigned ldsByteOff,
                                                   int cols, int rows,
                                                   int strideElems,
                                                   int padCode) {
  const unsigned long long ga = (unsigned long long)(const void*)gsrc;
  u32x4 g0;
  g0[0] = 1u;                                    // count = 1 (valid D#)
  g0[1] = ldsByteOff;                            // lds_addr
  g0[2] = (unsigned)ga;                          // global_addr[31:0]
  g0[3] = (unsigned)((ga >> 32) & 0x01FFFFFFull) // global_addr[56:32]
          | (2u << 30);                          // type = 2 ("image")
  i32x8 g1;
  g1[0] = (1 << 16)                              // data_size = 2 bytes
        | (1 << 20)                              // pad_enable
        | (padCode << 22)                        // pad_interval
        | (3 << 25);                             // pad_amount = 4 DWORDs (16B)
  g1[1] = cols << 16;                            // tensor_dim0[15:0] @ bit48
  g1[2] = rows << 16;                            // tensor_dim1[15:0] @ bit80
  g1[3] = cols << 16;                            // tile_dim0 @ bit112
  g1[4] = rows;                                  // tile_dim1 @ bit128
  g1[5] = strideElems;                           // tensor_dim0_stride[31:0]
  g1[6] = 0;
  g1[7] = 0;
  const i32x4 z4 = {0, 0, 0, 0};
  const i32x8 z8 = {0, 0, 0, 0, 0, 0, 0, 0};
  // 6-arg (clang-23 / therock-10.0) form: (g0, g1, g2, g3, g4, cpol).
  __builtin_amdgcn_tensor_load_to_lds(g0, g1, z4, z4, z8, 0);
}

static __device__ __forceinline__ v16bf ldfrag(const __bf16* base) {
  // ISA 16-bit A/B fragment: lane holds one row; two 16B chunks at
  // k = (lane>>4)*8 and k = (lane>>4)*8 + 16 (caller pre-offsets base).
  v16bf v;
  bfx8* h = reinterpret_cast<bfx8*>(&v);
  h[0] = *reinterpret_cast<const bfx8*>(base);
  h[1] = *reinterpret_cast<const bfx8*>(base + 16);
  return v;
}

static __device__ __forceinline__ bfx8 cvt8(const float4& f0, const float4& f1) {
  bfx8 bv;
  bv[0] = f2bf(f0.x); bv[1] = f2bf(f0.y); bv[2] = f2bf(f0.z); bv[3] = f2bf(f0.w);
  bv[4] = f2bf(f1.x); bv[5] = f2bf(f1.y); bv[6] = f2bf(f1.z); bv[7] = f2bf(f1.w);
  return bv;
}

// ---------------------------------------------------------------------------
// LayerNorm: 1 block per row of 768. fp32 in; fp32 and/or bf16 out.
// ---------------------------------------------------------------------------
__global__ __launch_bounds__(256)
void ln_kernel(const float* __restrict__ x, const float* __restrict__ gamma,
               const float* __restrict__ beta, float* __restrict__ outF,
               __bf16* __restrict__ outB) {
  const int row = blockIdx.x;
  const int t = threadIdx.x;
  const float* xr = x + (size_t)row * DMODEL;
  float v0 = xr[t], v1 = xr[t + 256], v2 = xr[t + 512];
  float s = v0 + v1 + v2;
  float q = v0 * v0 + v1 * v1 + v2 * v2;
#pragma unroll
  for (int m = 1; m < 32; m <<= 1) {
    s += __shfl_xor(s, m, 32);
    q += __shfl_xor(q, m, 32);
  }
  __shared__ float ss[8], sq[8];
  if ((t & 31) == 0) { ss[t >> 5] = s; sq[t >> 5] = q; }
  __syncthreads();
  float S = 0.f, Q = 0.f;
#pragma unroll
  for (int i = 0; i < 8; ++i) { S += ss[i]; Q += sq[i]; }
  const float mean = S * (1.f / DMODEL);
  const float var = Q * (1.f / DMODEL) - mean * mean;
  const float rs = rsqrtf(var + 1e-5f);
#pragma unroll
  for (int i = 0; i < 3; ++i) {
    const int c = t + i * 256;
    const float v = (i == 0) ? v0 : (i == 1 ? v1 : v2);
    const float y = (v - mean) * rs * gamma[c] + beta[c];
    if (outF) outF[(size_t)row * DMODEL + c] = y;
    if (outB) outB[(size_t)row * DMODEL + c] = f2bf(y);
  }
}

// ---------------------------------------------------------------------------
// GEMM: out[M,N] = A[M,K](bf16) @ W[N,K](f32->bf16 in LDS)^T + bias
//       (+ optional exact-erf GELU, + optional fp32 residual).
// Block tile 128x64, 8 waves of 32x32, K-step 32, WMMA f32_16x16x32_bf16.
// Double-buffered pipeline: TDM keeps two A-tile DMAs in flight
// (s_wait_tensorcnt 1); next W tile is loaded to VGPRs during current
// compute and converted/stored the following iteration.
// ---------------------------------------------------------------------------
__global__ __launch_bounds__(256)
void gemm_kernel(const __bf16* __restrict__ A, const float* __restrict__ W,
                 const float* __restrict__ bias, const float* __restrict__ resid,
                 float* __restrict__ outF, __bf16* __restrict__ outB,
                 int N, int K, int doGelu) {
  __shared__ __bf16 As[2][128 * 40];   // 128 rows x 32 k (+8 pad) -- TDM-filled
  __shared__ __bf16 Bs[2][64 * 40];    // 64 cols(n) x 32 k (+8 pad)
  const int t = threadIdx.x;
  const int lane = t & 31;
  const int w = t >> 5;
  const int wm = w >> 1;            // 0..3 -> M offset wm*32
  const int wn = w & 1;             // 0..1 -> N offset wn*32
  const int r = lane & 15;
  const int hb = (lane >> 4) * 8;

  const int m0 = blockIdx.y * 128;
  const int n0 = blockIdx.x * 64;
  const unsigned asOff0 = lds_off(As[0]);
  const unsigned asOff1 = lds_off(As[1]);

  v8f c00 = {}, c01 = {}, c10 = {}, c11 = {};

  const int brow = t >> 2;            // 0..63
  const int bcol = (t & 3) * 8;       // 0..24
  const float* wsrc = W + (size_t)(n0 + brow) * K + bcol;
  const __bf16* asrc = A + (size_t)m0 * K;

  // Prologue: start TDM for A tile 0; pull W tile 0 into registers.
  if (w == 0) tdm_load_2d(asrc, asOff0, 32, 128, K, /*padCode=*/3);
  float4 f0 = *reinterpret_cast<const float4*>(wsrc);
  float4 f1 = *reinterpret_cast<const float4*>(wsrc + 4);

  const int nk = K / 32;
  for (int kt = 0; kt < nk; ++kt) {
    const int cur = kt & 1;
    // bar1: all reads of buffer 'cur^1' (iteration kt-1) have completed,
    // so it is safe to start filling it below.
    __syncthreads();
    if (w == 0 && kt + 1 < nk)  // TDM for A tile kt+1 into the other buffer
      tdm_load_2d(asrc + (kt + 1) * 32, cur ? asOff0 : asOff1, 32, 128, K, 3);
    // Store staged W tile kt (native v_cvt bf16) into Bs[cur].
    *reinterpret_cast<bfx8*>(&Bs[cur][brow * 40 + bcol]) = cvt8(f0, f1);
    if (kt + 1 < nk) {  // issue global loads for W tile kt+1 (hide under WMMA)
      const float* src = wsrc + (kt + 1) * 32;
      f0 = *reinterpret_cast<const float4*>(src);
      f1 = *reinterpret_cast<const float4*>(src + 4);
      if (kt + 2 < nk)  // gfx1250 global_prefetch_b8 two tiles ahead
        __builtin_prefetch(src + 32, 0, 1);
    }
    // Drain only the older TDM: tile kt must be resident, kt+1 stays in flight.
    if (w == 0) {
      if (kt + 1 < nk) __builtin_amdgcn_s_wait_tensorcnt(1);
      else             __builtin_amdgcn_s_wait_tensorcnt(0);
    }
    __syncthreads();  // bar2: As[cur]/Bs[cur] visible to all waves

    const __bf16* as = As[cur];
    const __bf16* bs = Bs[cur];
    const v16bf a0 = ldfrag(as + (wm * 32 + r) * 40 + hb);
    const v16bf a1 = ldfrag(as + (wm * 32 + 16 + r) * 40 + hb);
    const v16bf b0 = ldfrag(bs + (wn * 32 + r) * 40 + hb);
    const v16bf b1 = ldfrag(bs + (wn * 32 + 16 + r) * 40 + hb);
    c00 = __builtin_amdgcn_wmma_f32_16x16x32_bf16(false, a0, false, b0, (short)0, c00, false, false);
    c01 = __builtin_amdgcn_wmma_f32_16x16x32_bf16(false, a0, false, b1, (short)0, c01, false, false);
    c10 = __builtin_amdgcn_wmma_f32_16x16x32_bf16(false, a1, false, b0, (short)0, c10, false, false);
    c11 = __builtin_amdgcn_wmma_f32_16x16x32_bf16(false, a1, false, b1, (short)0, c11, false, false);
  }

  // Epilogue. C layout: VGPR v -> row v (+8 for lanes 16-31 == hb), col = lane&15.
  const int wr = m0 + wm * 32;
  const int wc = n0 + wn * 32;
#pragma unroll
  for (int tile = 0; tile < 4; ++tile) {
    const v8f c = (tile == 0) ? c00 : (tile == 1) ? c01 : (tile == 2) ? c10 : c11;
    const int mb = wr + (tile >> 1) * 16;
    const int col = wc + (tile & 1) * 16 + r;
    const float bia = bias[col];
#pragma unroll
    for (int v = 0; v < 8; ++v) {
      const int row = mb + v + hb;
      float val = c[v] + bia;
      if (doGelu) val = 0.5f * val * (1.f + erff(val * 0.70710678118654752f));
      if (resid) val += resid[(size_t)row * N + col];
      if (outF) outF[(size_t)row * N + col] = val;
      if (outB) outB[(size_t)row * N + col] = f2bf(val);
    }
  }
}

// ---------------------------------------------------------------------------
// Flash attention: grid (SEQ/128, B*H). 8 waves x 16 query rows each.
// qkv[R,2304] bf16 -> out[R,768] bf16 (heads merged). Online softmax in fp32.
// K tile staged by TDM; V tile transposed through VGPRs.
// ---------------------------------------------------------------------------
__global__ __launch_bounds__(256)
void attn_kernel(const __bf16* __restrict__ qkv, __bf16* __restrict__ out) {
  __shared__ __bf16 Kt[32 * 72];        // 32 keys x 64 dh (+8 pad) -- TDM-filled
  __shared__ __bf16 Vt[64 * 40];        // transposed: 64 dh x 32 keys (+8 pad)
  __shared__ __bf16 Pb[8 * 16 * 40];    // per-wave P: 16 rows x 32 cols (+8 pad)
  const int t = threadIdx.x;
  const int lane = t & 31;
  const int w = t >> 5;
  const int r = lane & 15;
  const int hb = (lane >> 4) * 8;
  const int b = blockIdx.y / NHEAD;
  const int h = blockIdx.y % NHEAD;
  const int rb = b * SEQ;
  const int qr0 = blockIdx.x * 128 + w * 16;
  const size_t hoff = (size_t)h * DHEAD;
  const unsigned ktOff = lds_off(Kt);

  // Q fragments: 16x64 as two 16x32 A-frags, kept in registers.
  v16bf qa0, qa1;
  {
    const __bf16* qrow = qkv + (size_t)(rb + qr0 + r) * QKVW + hoff;
    bfx8* p0 = reinterpret_cast<bfx8*>(&qa0);
    p0[0] = *(const bfx8*)(qrow + hb);
    p0[1] = *(const bfx8*)(qrow + hb + 16);
    bfx8* p1 = reinterpret_cast<bfx8*>(&qa1);
    p1[0] = *(const bfx8*)(qrow + 32 + hb);
    p1[1] = *(const bfx8*)(qrow + 32 + hb + 16);
  }
  v8f o0 = {}, o1 = {}, o2 = {}, o3 = {};
  float mrow[8], lrow[8];
#pragma unroll
  for (int v = 0; v < 8; ++v) { mrow[v] = -1e30f; lrow[v] = 0.f; }

  const int ldr = t >> 3;        // 0..31 key row
  const int ldc = (t & 7) * 8;   // 0..56 dh col
  const float scale = 0.125f;    // 1/sqrt(64)

  for (int kb = 0; kb < SEQ; kb += 32) {
    __syncthreads();
    // K tile: TDM DMA of 32x64 bf16 (stride QKVW), pad 16B per 128B row.
    if (w == 0)
      tdm_load_2d(qkv + (size_t)(rb + kb) * QKVW + DMODEL + hoff, ktOff,
                  64, 32, QKVW, /*padCode=*/4);
    {  // V tile, transposed so B-frags for P@V read contiguous key chunks
      const __bf16* src = qkv + (size_t)(rb + kb + ldr) * QKVW + 2 * DMODEL + hoff + ldc;
      bfx8 tv = *(const bfx8*)src;
#pragma unroll
      for (int j = 0; j < 8; ++j) Vt[(ldc + j) * 40 + ldr] = tv[j];
    }
    if (w == 0) __builtin_amdgcn_s_wait_tensorcnt(0);
    __syncthreads();

    // S tiles (16 q-rows x 32 keys), fp32 accumulate over dh=64.
    v8f s0 = {}, s1 = {};
    {
      const v16bf k00 = ldfrag(Kt + (0 + r) * 72 + 0 + hb);
      const v16bf k01 = ldfrag(Kt + (0 + r) * 72 + 32 + hb);
      const v16bf k10 = ldfrag(Kt + (16 + r) * 72 + 0 + hb);
      const v16bf k11 = ldfrag(Kt + (16 + r) * 72 + 32 + hb);
      s0 = __builtin_amdgcn_wmma_f32_16x16x32_bf16(false, qa0, false, k00, (short)0, s0, false, false);
      s0 = __builtin_amdgcn_wmma_f32_16x16x32_bf16(false, qa1, false, k01, (short)0, s0, false, false);
      s1 = __builtin_amdgcn_wmma_f32_16x16x32_bf16(false, qa0, false, k10, (short)0, s1, false, false);
      s1 = __builtin_amdgcn_wmma_f32_16x16x32_bf16(false, qa1, false, k11, (short)0, s1, false, false);
    }

    // Online softmax. C layout: lanes 0-15 hold rows v, lanes 16-31 rows v+8;
    // xor masks < 16 reduce within each 16-lane half (independent rows).
    __bf16* prow_base = Pb + w * 16 * 40;
#pragma unroll
    for (int v = 0; v < 8; ++v) {
      const float x0 = s0[v] * scale;
      const float x1 = s1[v] * scale;
      float rm = fmaxf(x0, x1);
#pragma unroll
      for (int m = 1; m < 16; m <<= 1) rm = fmaxf(rm, __shfl_xor(rm, m, 32));
      const float mn = fmaxf(mrow[v], rm);
      const float al = __expf(mrow[v] - mn);
      const float p0 = __expf(x0 - mn);
      const float p1 = __expf(x1 - mn);
      float rs = p0 + p1;
#pragma unroll
      for (int m = 1; m < 16; m <<= 1) rs += __shfl_xor(rs, m, 32);
      lrow[v] = lrow[v] * al + rs;
      mrow[v] = mn;
      o0[v] *= al; o1[v] *= al; o2[v] *= al; o3[v] *= al;
      __bf16* pp = prow_base + (v + hb) * 40;   // C-layout -> row-major P
      pp[r] = f2bf(p0);
      pp[16 + r] = f2bf(p1);
    }
    __syncthreads();

    // O(16x64) += P(16x32) @ V(32x64): 4 WMMAs over dh tiles.
    v16bf pa;
    {
      const __bf16* base = prow_base + r * 40 + hb;
      bfx8* hp = reinterpret_cast<bfx8*>(&pa);
      hp[0] = *(const bfx8*)base;
      hp[1] = *(const bfx8*)(base + 16);
    }
#pragma unroll
    for (int j = 0; j < 4; ++j) {
      const __bf16* base = Vt + (j * 16 + r) * 40 + hb;
      v16bf vb;
      bfx8* hv = reinterpret_cast<bfx8*>(&vb);
      hv[0] = *(const bfx8*)base;
      hv[1] = *(const bfx8*)(base + 16);
      v8f& oo = (j == 0) ? o0 : (j == 1) ? o1 : (j == 2) ? o2 : o3;
      oo = __builtin_amdgcn_wmma_f32_16x16x32_bf16(false, pa, false, vb, (short)0, oo, false, false);
    }
  }

  // Normalize and write merged-head output (bf16).
#pragma unroll
  for (int v = 0; v < 8; ++v) {
    const float inv = 1.f / lrow[v];
    const int row = rb + qr0 + v + hb;
    __bf16* dst = out + (size_t)row * DMODEL + hoff;
    dst[0 + r]  = f2bf(o0[v] * inv);
    dst[16 + r] = f2bf(o1[v] * inv);
    dst[32 + r] = f2bf(o2[v] * inv);
    dst[48 + r] = f2bf(o3[v] * inv);
  }
}

// ---------------------------------------------------------------------------
// Host orchestration.
// ---------------------------------------------------------------------------
extern "C" void kernel_launch(void* const* d_in, const int* in_sizes, int n_in,
                              void* d_out, int out_size, void* d_ws, size_t ws_size,
                              hipStream_t stream) {
  (void)in_sizes; (void)n_in; (void)out_size; (void)ws_size;
  const float* x0   = (const float*)d_in[0];
  const float* Wqkv = (const float*)d_in[1];
  const float* bqkv = (const float*)d_in[2];
  const float* Wo   = (const float*)d_in[3];
  const float* bo   = (const float*)d_in[4];
  const float* g1   = (const float*)d_in[5];
  const float* be1  = (const float*)d_in[6];
  const float* g2   = (const float*)d_in[7];
  const float* be2  = (const float*)d_in[8];
  const float* W1   = (const float*)d_in[9];
  const float* b1   = (const float*)d_in[10];
  const float* W2   = (const float*)d_in[11];
  const float* b2   = (const float*)d_in[12];
  float* xf = (float*)d_out;  // running x lives in d_out

  char* ws = (char*)d_ws;
  size_t off = 0;
  auto carve = [&](size_t bytes) {
    void* p = ws + off;
    off += (bytes + 255) & ~(size_t)255;
    return p;
  };
  __bf16* qkvb = (__bf16*)carve((size_t)RTOT * QKVW * 2);    // 18.9 MB
  __bf16* hb   = (__bf16*)carve((size_t)RTOT * DFF * 2);     // 25.2 MB
  __bf16* ab   = (__bf16*)carve((size_t)RTOT * DMODEL * 2);  //  6.3 MB
  __bf16* actb = (__bf16*)carve((size_t)RTOT * DMODEL * 2);  //  6.3 MB
  float*  x1f  = (float*)carve((size_t)RTOT * DMODEL * 4);   // 12.6 MB

  const dim3 blk(256);
  const dim3 gLN(RTOT);
  const dim3 gQKV(QKVW / 64, RTOT / 128);
  const dim3 gD(DMODEL / 64, RTOT / 128);
  const dim3 gFF(DFF / 64, RTOT / 128);
  const dim3 gAT(SEQ / 128, BATCH * NHEAD);

  const float* xin = x0;
  for (int l = 0; l < LAYERS; ++l) {
    const size_t ld = (size_t)l * DMODEL;
    // src = LN(x, g1, b1) -> bf16
    ln_kernel<<<gLN, blk, 0, stream>>>(xin, g1 + ld, be1 + ld, nullptr, actb);
    // qkv = src @ Wqkv^T + bqkv -> bf16
    gemm_kernel<<<gQKV, blk, 0, stream>>>(actb, Wqkv + (size_t)l * QKVW * DMODEL,
                                          bqkv + (size_t)l * QKVW, nullptr, nullptr,
                                          qkvb, QKVW, DMODEL, 0);
    // attention (flash, per-head) -> bf16 merged heads
    attn_kernel<<<gAT, blk, 0, stream>>>(qkvb, ab);
    // x1 = x + attn @ Wo^T + bo -> f32
    gemm_kernel<<<gD, blk, 0, stream>>>(ab, Wo + (size_t)l * DMODEL * DMODEL,
                                        bo + ld, xin, x1f, nullptr, DMODEL, DMODEL, 0);
    // x = LN(x1, g1, b1) -> f32 (d_out) + bf16
    ln_kernel<<<gLN, blk, 0, stream>>>(x1f, g1 + ld, be1 + ld, xf, actb);
    // h = gelu(x @ W1^T + b1) -> bf16
    gemm_kernel<<<gFF, blk, 0, stream>>>(actb, W1 + (size_t)l * DFF * DMODEL,
                                         b1 + (size_t)l * DFF, nullptr, nullptr,
                                         hb, DFF, DMODEL, 1);
    // x2 = x + h @ W2^T + b2 -> f32
    gemm_kernel<<<gD, blk, 0, stream>>>(hb, W2 + (size_t)l * DMODEL * DFF,
                                        b2 + ld, xf, x1f, nullptr, DMODEL, DFF, 0);
    // x = LN(x2, g2, b2) -> f32 (d_out)
    ln_kernel<<<gLN, blk, 0, stream>>>(x1f, g2 + ld, be2 + ld, xf, nullptr);
    xin = xf;
  }
}
